// LTPBlock_43087111913968
// MI455X (gfx1250) — matrix operand order repaired and checked by reading, not compiled, not run on hardware
//
#include <hip/hip_runtime.h>
#include <math.h>

// ---------------- problem constants ----------------
#define B_   64
#define N_   197
#define C_   768
#define H_   12
#define D_   64
#define M_   (B_ * N_)      // 12608 rows
#define C3_  (3 * C_)       // 2304
#define C4_  (4 * C_)       // 3072
#define QT_  13             // ceil(197/16) query/key tiles
#define SCALE_   0.125f     // D^-0.5
#define EPS_SM_  1e-6f
#define LN_EPS_  1e-5f

typedef __attribute__((ext_vector_type(16))) __bf16 v16bf;
typedef __attribute__((ext_vector_type(8)))  float  v8f;

// fragment <-> 2x16B view (per-lane data is two contiguous 16B chunks)
union FragU { uint4 q[2]; v16bf v; };

// ---------------------------------------------------------------------------
// CDNA5 async copy: global -> LDS, 16B per lane, tracked on ASYNCcnt.
// (ISA 15.18.3 opcode 98 GLOBAL_LOAD_ASYNC_TO_LDS_B128; VDST = LDS byte addr)
// ---------------------------------------------------------------------------
__device__ __forceinline__ void async_ld_b128(const void* gptr, void* lds) {
    unsigned l = (unsigned)(uintptr_t)lds;   // low 32 bits of generic = LDS offset
    asm volatile("global_load_async_to_lds_b128 %0, %1, off"
                 :: "v"(l), "v"(gptr) : "memory");
}
__device__ __forceinline__ void wait_async_le2() {
    asm volatile("s_wait_asynccnt 0x2" ::: "memory");
}
__device__ __forceinline__ void wait_async_0() {
    asm volatile("s_wait_asynccnt 0x0" ::: "memory");
}

// ---------------------------------------------------------------------------
// f32 -> bf16 conversion
// ---------------------------------------------------------------------------
__global__ __launch_bounds__(256) void cvt_bf16_kernel(const float* __restrict__ in,
                                                       __bf16* __restrict__ out, int n) {
    int i = blockIdx.x * 256 + threadIdx.x;
    if (i < n) out[i] = (__bf16)in[i];
}

// f32 [K x Ntot] -> bf16 transposed [Ntot x K] (one-time; weights stay in L2)
__global__ __launch_bounds__(256) void cvt_transpose_kernel(const float* __restrict__ in,
                                                            __bf16* __restrict__ out,
                                                            int K, int Ntot) {
    int i = blockIdx.x * 256 + threadIdx.x;
    if (i >= K * Ntot) return;
    int k = i / Ntot;
    int n = i - k * Ntot;
    out[(size_t)n * K + k] = (__bf16)in[i];
}

// ---------------------------------------------------------------------------
// LayerNorm over C=768, one block per row, bf16 output
// ---------------------------------------------------------------------------
__global__ __launch_bounds__(256) void ln_kernel(const float* __restrict__ x,
                                                 const float* __restrict__ w,
                                                 const float* __restrict__ bv,
                                                 __bf16* __restrict__ out) {
    __shared__ float red[256];
    const int row = blockIdx.x;
    const int tid = threadIdx.x;
    const float* xr = x + (size_t)row * C_;

    float s = 0.f;
    for (int c = tid; c < C_; c += 256) s += xr[c];
    red[tid] = s; __syncthreads();
    for (int off = 128; off > 0; off >>= 1) {
        if (tid < off) red[tid] += red[tid + off];
        __syncthreads();
    }
    const float mean = red[0] * (1.0f / C_);
    __syncthreads();

    float s2 = 0.f;
    for (int c = tid; c < C_; c += 256) { float d = xr[c] - mean; s2 += d * d; }
    red[tid] = s2; __syncthreads();
    for (int off = 128; off > 0; off >>= 1) {
        if (tid < off) red[tid] += red[tid + off];
        __syncthreads();
    }
    const float inv = rsqrtf(red[0] * (1.0f / C_) + LN_EPS_);

    __bf16* orow = out + (size_t)row * C_;
    for (int c = tid; c < C_; c += 256)
        orow[c] = (__bf16)(((xr[c] - mean) * inv) * w[c] + bv[c]);
}

// ---------------------------------------------------------------------------
// bf16 WMMA GEMM: C[M x Ntot] = A[M x K] * W^T[Ntot x K] (+epilogue)
// Block tile 128x128, 8 waves, each wave 32x64 (2x4 wmma frags), K step 32.
// A tile staged via async-to-LDS (double buffered on ASYNCcnt);
// B fragments read directly from pre-transposed bf16 weights (L2-resident).
// EPI 0: +bias -> bf16 out   (qkv)
// EPI 1: +bias+resid -> f32  (proj, fc2)
// EPI 2: +bias, GELU -> bf16 (fc1)
// ---------------------------------------------------------------------------
template <int EPI>
__global__ __launch_bounds__(256) void gemm_bf16_kernel(
        const __bf16* __restrict__ A, const __bf16* __restrict__ Wt,
        const float* __restrict__ bias, const float* __restrict__ resid,
        float* __restrict__ outf, __bf16* __restrict__ outh,
        int M, int Ntot, int K) {
    __shared__ __attribute__((aligned(16))) __bf16 As[2][128][40]; // 80B row stride

    const int tid   = threadIdx.x;
    const int lane  = tid & 31;
    const int w     = tid >> 5;
    const int wm    = w >> 1;        // 0..3  -> M offset
    const int wn    = w & 1;         // 0..1  -> N offset
    const int rlane = lane & 15;
    const int half  = lane >> 4;

    const int m0 = blockIdx.y * 128;
    const int n0 = blockIdx.x * 128;

    v8f acc[2][4] = {};

    auto issueA = [&](int kt) {
        const int buf = (kt >> 5) & 1;
        #pragma unroll
        for (int i = 0; i < 2; ++i) {
            int vtx = tid + i * 256;
            int row = vtx >> 2;            // 4 x 16B chunks per 32-wide row
            int k8  = (vtx & 3) * 8;
            int gr  = m0 + row; if (gr >= M) gr = M - 1;
            async_ld_b128(A + (size_t)gr * K + kt + k8, &As[buf][row][k8]);
        }
    };

    issueA(0);
    const int nk = K >> 5;
    for (int it = 0; it < nk; ++it) {
        const int buf = it & 1;
        const int kt  = it << 5;
        if (it + 1 < nk) { issueA(kt + 32); wait_async_le2(); }
        else             { wait_async_0(); }
        __syncthreads();

        // A fragments: two contiguous 16B chunks per lane (ISA 7.12.2 layout)
        FragU a0, a1;
        {
            const int r0 = wm * 32 + rlane;
            a0.q[0] = *(const uint4*)&As[buf][r0][half * 8];
            a0.q[1] = *(const uint4*)&As[buf][r0][16 + half * 8];
            a1.q[0] = *(const uint4*)&As[buf][r0 + 16][half * 8];
            a1.q[1] = *(const uint4*)&As[buf][r0 + 16][16 + half * 8];
        }
        #pragma unroll
        for (int j = 0; j < 4; ++j) {
            FragU bfr;
            const int col = n0 + wn * 64 + j * 16 + rlane;
            const __bf16* bp = Wt + (size_t)col * K + kt + half * 8;
            bfr.q[0] = *(const uint4*)bp;
            bfr.q[1] = *(const uint4*)(bp + 16);
            acc[0][j] = __builtin_amdgcn_wmma_f32_16x16x32_bf16(
                false, a0.v, false, bfr.v, (short)0, acc[0][j], false, false);
            acc[1][j] = __builtin_amdgcn_wmma_f32_16x16x32_bf16(
                false, a1.v, false, bfr.v, (short)0, acc[1][j], false, false);
        }
        __syncthreads();
    }

    // epilogue: C/D layout: VGPR r -> M = r + (lane>=16 ? 8:0), N = lane&15
    #pragma unroll
    for (int i = 0; i < 2; ++i)
        #pragma unroll
        for (int j = 0; j < 4; ++j) {
            const int col = n0 + wn * 64 + j * 16 + rlane;
            #pragma unroll
            for (int r = 0; r < 8; ++r) {
                const int row = m0 + wm * 32 + i * 16 + r + half * 8;
                if (row < M) {
                    float v = acc[i][j][r] + bias[col];
                    if (EPI == 1) {
                        v += resid[(size_t)row * Ntot + col];
                        outf[(size_t)row * Ntot + col] = v;
                    } else if (EPI == 2) {
                        v = 0.5f * v * (1.0f + erff(v * 0.70710678118654752f));
                        outh[(size_t)row * Ntot + col] = (__bf16)v;
                    } else {
                        outh[(size_t)row * Ntot + col] = (__bf16)v;
                    }
                }
            }
        }
}

// ---------------------------------------------------------------------------
// Attention: block = 2 waves, one (b,h); each wave one 16-query tile.
// V panel async-copied to LDS (overlapped with QK^T + softmax on ASYNCcnt).
// qkv layout: [b, n, 3C] bf16 ; output attn_out: [b, n, C] bf16
// ---------------------------------------------------------------------------
__global__ __launch_bounds__(64) void attn_kernel(
        const __bf16* __restrict__ qkv, const float* __restrict__ mask,
        __bf16* __restrict__ attn_out, float* __restrict__ imp_part) {
    constexpr int NP = 224;                       // padded key dim (7 * 32)
    __shared__ float S[2][16][NP];
    __shared__ __attribute__((aligned(16))) __bf16 Vs[NP][64];

    const int tid   = threadIdx.x;
    const int wave  = tid >> 5;
    const int lane  = tid & 31;
    const int rlane = lane & 15;
    const int half  = lane >> 4;

    const int bh = blockIdx.x;
    const int b  = bh / H_;
    const int h  = bh % H_;
    const int qt = blockIdx.y * 2 + wave;          // 0..13 (valid < 13)
    const int qn0 = qt * 16;

    float (*Sw)[NP] = S[wave];
    const size_t rs = (size_t)C3_;
    const __bf16* qbase = qkv + (size_t)b * N_ * rs + h * D_;
    const __bf16* kbase = qbase + C_;
    const __bf16* vbase = qbase + 2 * C_;

    // ---- kick off async V panel copy (224 x 64 bf16 -> LDS) --------------
    for (int i = tid; i < (NP * 64) / 8; i += 64) {
        int key = i >> 3;                // 8 x 16B chunks per 64-wide key row
        int d8  = (i & 7) * 8;
        int kc  = min(key, N_ - 1);
        async_ld_b128(vbase + (size_t)kc * rs + d8, &Vs[key][d8]);
    }

    // ---- phase 1: S = scale * Q K^T (overlaps with async V copy) ---------
    const int qrow = min(qn0 + rlane, N_ - 1);
    const __bf16* qrp = qbase + (size_t)qrow * rs;
    FragU qa[2];
    #pragma unroll
    for (int c = 0; c < 2; ++c) {
        qa[c].q[0] = *(const uint4*)(qrp + c * 32 + half * 8);
        qa[c].q[1] = *(const uint4*)(qrp + c * 32 + 16 + half * 8);
    }

    for (int kt = 0; kt < QT_; ++kt) {
        v8f acc = {};
        const int key = min(kt * 16 + rlane, N_ - 1);
        const __bf16* krp = kbase + (size_t)key * rs;
        #pragma unroll
        for (int c = 0; c < 2; ++c) {
            FragU kf;
            kf.q[0] = *(const uint4*)(krp + c * 32 + half * 8);
            kf.q[1] = *(const uint4*)(krp + c * 32 + 16 + half * 8);
            acc = __builtin_amdgcn_wmma_f32_16x16x32_bf16(
                false, qa[c].v, false, kf.v, (short)0, acc, false, false);
        }
        #pragma unroll
        for (int r = 0; r < 8; ++r)
            Sw[r + half * 8][kt * 16 + rlane] = acc[r] * SCALE_;
    }
    // zero the key padding (cols 197..223)
    for (int r = 0; r < 16; ++r)
        for (int m = 197 + lane; m < NP; m += 32) Sw[r][m] = 0.f;
    __syncthreads();

    // ---- phase 2: masked softmax per query row ---------------------------
    if (lane < 16) {
        const int row = lane;
        const int qn  = qn0 + row;
        if (qn < N_) {
            float mx = -3.402823466e+38f;
            for (int m = 0; m < N_; ++m) mx = fmaxf(mx, Sw[row][m]);
            float sum = 0.f;
            for (int m = 0; m < N_; ++m) {
                float wv = __expf(Sw[row][m] - mx) * mask[b * N_ + m];
                Sw[row][m] = wv;
                sum += wv;
            }
            const float inv = 1.0f / (sum + EPS_SM_);
            for (int m = 0; m < N_; ++m) Sw[row][m] *= inv;
        } else {
            for (int m = 0; m < N_; ++m) Sw[row][m] = 0.f;
        }
    }
    wait_async_0();                 // V panel now resident in LDS
    __syncthreads();

    // ---- phase 3a: deterministic importance partials ---------------------
    if (qt < QT_) {
        float* pp = imp_part + ((size_t)bh * QT_ + qt) * N_;
        for (int m = lane; m < N_; m += 32) {
            float s = 0.f;
            #pragma unroll
            for (int r = 0; r < 16; ++r) s += Sw[r][m];
            pp[m] = s;
        }
    }

    // ---- phase 3b: O = attn @ V ------------------------------------------
    v8f accd[4] = {};
    for (int kt = 0; kt < 7; ++kt) {                  // 7 * 32 = 224 keys
        v16bf af;
        #pragma unroll
        for (int j = 0; j < 8; ++j) {
            int kb  = (j >> 2) * 16 + half * 8 + (j & 3) * 2;
            int key = kt * 32 + kb;
            af[2 * j]     = (__bf16)Sw[rlane][key];
            af[2 * j + 1] = (__bf16)Sw[rlane][key + 1];
        }
        #pragma unroll
        for (int dt = 0; dt < 4; ++dt) {
            v16bf vf;
            const int d = dt * 16 + rlane;
            #pragma unroll
            for (int j = 0; j < 8; ++j) {
                int k0 = kt * 32 + (j >> 2) * 16 + half * 8 + (j & 3) * 2;
                vf[2 * j]     = Vs[k0][d];
                vf[2 * j + 1] = Vs[k0 + 1][d];
            }
            accd[dt] = __builtin_amdgcn_wmma_f32_16x16x32_bf16(
                false, af, false, vf, (short)0, accd[dt], false, false);
        }
    }
    #pragma unroll
    for (int dt = 0; dt < 4; ++dt)
        #pragma unroll
        for (int r = 0; r < 8; ++r) {
            const int qn = qn0 + r + half * 8;
            if (qn < N_)
                attn_out[((size_t)b * N_ + qn) * C_ + h * D_ + dt * 16 + rlane] =
                    (__bf16)accd[dt][r];
        }
}

// ---------------------------------------------------------------------------
// Mask finalize: reduce imp partials, apply inf@token0, threshold, where()
// ---------------------------------------------------------------------------
__global__ __launch_bounds__(256) void mask_kernel(
        const float* __restrict__ imp_part, const float* __restrict__ mask,
        const float* __restrict__ thr, float* __restrict__ mask_out) {
    const int i = blockIdx.x * 256 + threadIdx.x;
    if (i >= M_) return;
    const int b = i / N_;
    const int m = i % N_;
    const float* pb = imp_part + (size_t)b * H_ * QT_ * N_;
    float s = 0.f;
    for (int t = 0; t < H_ * QT_; ++t) s += pb[(size_t)t * N_ + m];
    float mean = s * (1.0f / (float)(H_ * N_));
    if (m == 0) mean = __builtin_inff();
    const float nm = (mean > thr[0]) ? 1.0f : 0.0f;
    const float mv = mask[i];
    mask_out[i] = (mv > 0.f) ? nm : mv;
}

// ---------------------------------------------------------------------------
// host launcher
// ---------------------------------------------------------------------------
extern "C" void kernel_launch(void* const* d_in, const int* in_sizes, int n_in,
                              void* d_out, int out_size, void* d_ws, size_t ws_size,
                              hipStream_t stream) {
    const float* x      = (const float*)d_in[0];
    const float* mask   = (const float*)d_in[1];
    const float* ln1_w  = (const float*)d_in[2];
    const float* ln1_b  = (const float*)d_in[3];
    const float* qkv_w  = (const float*)d_in[4];
    const float* qkv_b  = (const float*)d_in[5];
    const float* proj_w = (const float*)d_in[6];
    const float* proj_b = (const float*)d_in[7];
    const float* ln2_w  = (const float*)d_in[8];
    const float* ln2_b  = (const float*)d_in[9];
    const float* fc1_w  = (const float*)d_in[10];
    const float* fc1_b  = (const float*)d_in[11];
    const float* fc2_w  = (const float*)d_in[12];
    const float* fc2_b  = (const float*)d_in[13];
    const float* thr    = (const float*)d_in[14];

    char* ws = (char*)d_ws;
    size_t off = 0;
    auto take = [&](size_t bytes) -> char* {
        char* p = ws + off;
        off += (bytes + 255) & ~(size_t)255;
        return p;
    };

    __bf16* wqkv  = (__bf16*)take((size_t)C_ * C3_ * 2);   // [2304][768]  (W^T)
    __bf16* wproj = (__bf16*)take((size_t)C_ * C_  * 2);   // [768][768]   (W^T)
    __bf16* wfc1  = (__bf16*)take((size_t)C_ * C4_ * 2);   // [3072][768]  (W^T)
    __bf16* wfc2  = (__bf16*)take((size_t)C4_ * C_ * 2);   // [768][3072]  (W^T)
    __bf16* hbuf  = (__bf16*)take((size_t)M_ * C_  * 2);   // LN1 out, reused for LN2
    __bf16* qkvb  = (__bf16*)take((size_t)M_ * C3_ * 2);
    __bf16* aobuf = (__bf16*)take((size_t)M_ * C_  * 2);
    float*  x1    = (float*) take((size_t)M_ * C_  * 4);
    __bf16* gbuf  = (__bf16*)take((size_t)M_ * C4_ * 2);
    float*  imp   = (float*) take((size_t)B_ * H_ * QT_ * N_ * 4);

    float* out_x    = (float*)d_out;
    float* out_mask = out_x + (size_t)M_ * C_;

    const int MB = (M_ + 127) / 128;   // 99 row tiles

    // 1) weights -> transposed bf16 (then L2-resident)
    cvt_transpose_kernel<<<((C_*C3_)+255)/256, 256, 0, stream>>>(qkv_w,  wqkv,  C_, C3_);
    cvt_transpose_kernel<<<((C_*C_ )+255)/256, 256, 0, stream>>>(proj_w, wproj, C_, C_);
    cvt_transpose_kernel<<<((C_*C4_)+255)/256, 256, 0, stream>>>(fc1_w,  wfc1,  C_, C4_);
    cvt_transpose_kernel<<<((C4_*C_)+255)/256, 256, 0, stream>>>(fc2_w,  wfc2,  C4_, C_);

    // 2) LN1
    ln_kernel<<<M_, 256, 0, stream>>>(x, ln1_w, ln1_b, hbuf);

    // 3) qkv = h @ Wqkv + b            (12608 x 2304, K=768)
    gemm_bf16_kernel<0><<<dim3(C3_/128, MB), 256, 0, stream>>>(
        hbuf, wqkv, qkv_b, nullptr, nullptr, qkvb, M_, C3_, C_);

    // 4) attention (+ importance partials)
    attn_kernel<<<dim3(B_*H_, (QT_ + 1) / 2), 64, 0, stream>>>(qkvb, mask, aobuf, imp);

    // 5) x1 = x + attn_out @ Wproj + b (12608 x 768, K=768)
    gemm_bf16_kernel<1><<<dim3(C_/128, MB), 256, 0, stream>>>(
        aobuf, wproj, proj_b, x, x1, nullptr, M_, C_, C_);

    // 6) LN2
    ln_kernel<<<M_, 256, 0, stream>>>(x1, ln2_w, ln2_b, hbuf);

    // 7) g = gelu(h2 @ Wfc1 + b)       (12608 x 3072, K=768)
    gemm_bf16_kernel<2><<<dim3(C4_/128, MB), 256, 0, stream>>>(
        hbuf, wfc1, fc1_b, nullptr, nullptr, gbuf, M_, C4_, C_);

    // 8) out = x1 + g @ Wfc2 + b       (12608 x 768, K=3072)
    gemm_bf16_kernel<1><<<dim3(C_/128, MB), 256, 0, stream>>>(
        gbuf, wfc2, fc2_b, x1, out_x, nullptr, M_, C_, C4_);

    // 9) mask output
    mask_kernel<<<(M_+255)/256, 256, 0, stream>>>(imp, mask, thr, out_mask);
}
